// Quantizer_20650202759185
// MI455X (gfx1250) — matrix operand order repaired and checked by reading, not compiled, not run on hardware
//
#include <hip/hip_runtime.h>
#include <float.h>

typedef __attribute__((ext_vector_type(16))) _Float16 v16h;
typedef __attribute__((ext_vector_type(8)))  _Float16 v8h;
typedef __attribute__((ext_vector_type(8)))  float    v8f;
typedef __attribute__((ext_vector_type(4)))  unsigned int u32x4;
typedef __attribute__((ext_vector_type(8)))  int      i32x8;
typedef __attribute__((ext_vector_type(4)))  int      i32x4;

#define EMBED_DIM 256
#define NUM_EMBED 8192
#define TOKENS    32768
#define TOK_TILE  64                     // tokens per workgroup
#define CODE_TILE 64                     // codes per K-chunk
#define WAVES     4
#define BLOCK     (WAVES * 32)
#define N_CHUNKS  (NUM_EMBED / CODE_TILE)
#define CHUNK_ELEMS (CODE_TILE * EMBED_DIM)

// ---------------------------------------------------------------------------
// Pass 1: codebook f32 -> f16, plus ||e||^2 per code row.
// ---------------------------------------------------------------------------
__global__ __launch_bounds__(EMBED_DIM)
void vq_prep_kernel(const float* __restrict__ cb,
                    _Float16* __restrict__ cbH,
                    float* __restrict__ cbNorm) {
    const int row = blockIdx.x;
    const int tid = threadIdx.x;
    float v = cb[(size_t)row * EMBED_DIM + tid];
    cbH[(size_t)row * EMBED_DIM + tid] = (_Float16)v;

    __shared__ float red[EMBED_DIM];
    red[tid] = v * v;
    __syncthreads();
#pragma unroll
    for (int s = EMBED_DIM / 2; s > 0; s >>= 1) {
        if (tid < s) red[tid] += red[tid + s];
        __syncthreads();
    }
    if (tid == 0) cbNorm[row] = red[0];
}

// ---------------------------------------------------------------------------
// TDM descriptor: 2D tile, 64 rows x 256 f16, contiguous (stride 256),
// DMA'd from global into LDS at lds_byte_off.  ISA cdna5 §8.3/8.4.
// ---------------------------------------------------------------------------
__device__ __forceinline__ void tdm_load_b_chunk(unsigned int lds_byte_off,
                                                 const _Float16* gsrc) {
    unsigned long long ga = (unsigned long long)(uintptr_t)gsrc;
    u32x4 g0 = {};
    g0.x = 1u;                                    // count=1, user descriptor
    g0.y = lds_byte_off;                          // lds_addr [63:32]
    g0.z = (unsigned int)ga;                      // global_addr[31:0]
    g0.w = (unsigned int)((ga >> 32) & 0x01FFFFFFu) | (2u << 30); // addr[56:32], type=2
    i32x8 g1 = {};
    g1[0] = 0x00010000;                           // workgroup_mask=0, data_size=1 (2B)
    g1[1] = (int)(256u << 16);                    // tensor_dim0 = 256 (bits 79:48 lo16)
    g1[2] = (int)((unsigned)CODE_TILE << 16);     // tensor_dim1 = 64  (bits 111:80 lo16)
    g1[3] = (int)(256u << 16);                    // tile_dim0 = 256   (bits 127:112)
    g1[4] = CODE_TILE;                            // tile_dim1 = 64    (bits 143:128)
    g1[5] = 256;                                  // tensor_dim0_stride = 256 (bits 207:160)
    i32x4 gz4 = {};
    i32x8 gz8 = {};
    // 6-arg toolchain variant: (g0, g1, g2, g3, g4, cpol)
    __builtin_amdgcn_tensor_load_to_lds(g0, g1, gz4, gz4, gz8, 0);
}

// ---------------------------------------------------------------------------
// Pass 2: fused distance-GEMM (WMMA f16->f32) + running argmin + gather.
// dist(t,k) = ||e_k||^2 - 2 <x_t, e_k>   (||x_t||^2 dropped: argmin-invariant)
// B chunks are DMA'd by the Tensor Data Mover, double-buffered in LDS.
// ---------------------------------------------------------------------------
__global__ __launch_bounds__(BLOCK)
void vq_main_kernel(const float* __restrict__ input,      // NCHW f32
                    const _Float16* __restrict__ cbH,     // [K][C] f16
                    const float* __restrict__ cbNorm,     // [K]
                    const float* __restrict__ cbF32,      // [K][C] f32
                    float* __restrict__ out) {
    __shared__ __align__(128) _Float16 shA[TOK_TILE * EMBED_DIM];   // 32 KB
    __shared__ __align__(128) _Float16 shB[2][CHUNK_ELEMS];         // 64 KB ping/pong

    const int tid  = threadIdx.x;
    const int w    = tid >> 5;       // wave id 0..3  -> M sub-tile
    const int lane = tid & 31;
    const int half = lane >> 4;      // 16-lane half within wave32
    const int l15  = lane & 15;

    const int tBase = blockIdx.x * TOK_TILE;   // 64-aligned -> single image
    const int nImg  = tBase >> 10;             // H*W = 1024 tokens per image
    const int sBase = tBase & 1023;
    const float* inBase = input + (size_t)nImg * EMBED_DIM * 1024 + sBase;

    // Generic LDS pointer low 32 bits == LDS byte offset (flat aperture rule).
    const unsigned int ldsB0 = (unsigned int)(uintptr_t)&shB[0][0];
    const unsigned int ldsB1 = (unsigned int)(uintptr_t)&shB[1][0];

    // Wave 0 kicks off the TDM DMA of chunk 0 while everyone stages A.
    if (w == 0) tdm_load_b_chunk(ldsB0, cbH);

    // Stage A: fused NCHW->[token][channel] transpose + f32->f16 convert.
    for (int e = tid; e < TOK_TILE * EMBED_DIM; e += BLOCK) {
        int c = e >> 6;
        int t = e & 63;
        shA[t * EMBED_DIM + c] = (_Float16)inBase[c * 1024 + t];
    }

    float minVal[8];
    int   minIdx[8];
#pragma unroll
    for (int r = 0; r < 8; ++r) { minVal[r] = FLT_MAX; minIdx[r] = 0; }

    // A-matrix 16x32 f16 layout used here (contiguous k): lane m = l15;
    //   elems j -> K = ks + 16*half + j
    const _Float16* aRow = shA + (16 * w + l15) * EMBED_DIM;

    for (int kc = 0; kc < N_CHUNKS; ++kc) {
        if (w == 0) {
            if (kc + 1 < N_CHUNKS) {
                // Prefetch next chunk into the other buffer (overlaps compute),
                // then wait until only that prefetch is still outstanding:
                // TDM ops of one wave complete in order => chunk kc has landed.
                tdm_load_b_chunk((kc & 1) ? ldsB0 : ldsB1,
                                 cbH + (size_t)(kc + 1) * CHUNK_ELEMS);
                __builtin_amdgcn_s_wait_tensorcnt(1);
            } else {
                __builtin_amdgcn_s_wait_tensorcnt(0);
            }
        }
        __syncthreads();   // chunk kc visible to all waves

        const _Float16* bBuf = shB[kc & 1];
#pragma unroll
        for (int nt = 0; nt < CODE_TILE / 16; ++nt) {
            // B-matrix 32x16 layout: lane n = l15; elems j -> K = ks + 16*half + j
            const _Float16* bRow = bBuf + (nt * 16 + l15) * EMBED_DIM;
            v8f acc = {};
#pragma unroll
            for (int ks = 0; ks < EMBED_DIM; ks += 32) {
                v8h aLo = *(const v8h*)(aRow + ks + 16 * half);
                v8h aHi = *(const v8h*)(aRow + ks + 16 * half + 8);
                v16h b  = *(const v16h*)(bRow + ks + 16 * half);
                v16h a;
#pragma unroll
                for (int i = 0; i < 8; ++i) { a[i] = aLo[i]; a[i + 8] = aHi[i]; }
                acc = __builtin_amdgcn_wmma_f32_16x16x32_f16(
                    false, a, false, b, (short)0, acc, false, false);
            }
            // C/D layout: N = l15 (constant per lane), M = r + 8*half
            const int n = kc * CODE_TILE + nt * 16 + l15;
            const float nrm = cbNorm[n];
#pragma unroll
            for (int r = 0; r < 8; ++r) {
                float dist = nrm - 2.0f * acc[r];
                if (dist < minVal[r]) { minVal[r] = dist; minIdx[r] = n; }
            }
        }
        __syncthreads();   // all waves done with buf[kc&1] before it is re-DMA'd
    }

    int* sBest = (int*)&shB[0][0];   // reuse LDS for the 64 winning indices

    // Cross-lane argmin within each 16-lane half (xor masks 1..8 stay in-half).
#pragma unroll
    for (int r = 0; r < 8; ++r) {
        float v  = minVal[r];
        int   ix = minIdx[r];
#pragma unroll
        for (int m = 8; m >= 1; m >>= 1) {
            float ov = __shfl_xor(v, m, 32);
            int   oi = __shfl_xor(ix, m, 32);
            if (ov < v || (ov == v && oi < ix)) { v = ov; ix = oi; }
        }
        if (l15 == 0) sBest[16 * w + 8 * half + r] = ix;   // token = 16w+8*half+r
    }
    __syncthreads();

    // Outputs: quantize | quantize_detach (identical) | data, each [T][C] f32.
    float* outQ  = out;
    float* outQd = out + (size_t)TOKENS * EMBED_DIM;
    float* outD  = out + 2 * (size_t)TOKENS * EMBED_DIM;
    for (int e = tid; e < TOK_TILE * EMBED_DIM; e += BLOCK) {
        int t = e >> 8;              // channel-fast => coalesced stores
        int c = e & 255;
        int bi = sBest[t];
        float q = cbF32[(size_t)bi * EMBED_DIM + c];
        float d = inBase[c * 1024 + t];   // strided but L2/WGP$-resident tile
        size_t o = (size_t)(tBase + t) * EMBED_DIM + c;
        outQ[o]  = q;
        outQd[o] = q;
        outD[o]  = d;
    }
}

// ---------------------------------------------------------------------------
extern "C" void kernel_launch(void* const* d_in, const int* in_sizes, int n_in,
                              void* d_out, int out_size, void* d_ws, size_t ws_size,
                              hipStream_t stream) {
    const float* input = (const float*)d_in[0];   // (32, 256, 32, 32) f32
    const float* cb    = (const float*)d_in[1];   // (8192, 256) f32

    // Workspace: [0, 4MB) codebook f16, [4MB, 4MB+32KB) code norms.
    _Float16* cbH   = (_Float16*)d_ws;
    float*   cbNorm = (float*)((char*)d_ws + (size_t)NUM_EMBED * EMBED_DIM * sizeof(_Float16));
    float*   out    = (float*)d_out;

    vq_prep_kernel<<<NUM_EMBED, EMBED_DIM, 0, stream>>>(cb, cbH, cbNorm);
    vq_main_kernel<<<TOKENS / TOK_TILE, BLOCK, 0, stream>>>(input, cbH, cbNorm, cb, out);
}
